// STM_head_55602646614367
// MI455X (gfx1250) — matrix operand order repaired and checked by reading, compiled, not run on hardware
//
#include <hip/hip_runtime.h>
#include <hip/hip_bf16.h>

// ---------------------------------------------------------------------------
// STM attention head for MI455X (gfx1250): bf16 WMMA, online softmax over q,
// LDS-resident probability strip (231KB -- needs CDNA5's 320KB WGP LDS),
// 32-wide q blocking to halve L2 traffic on the value matrix.
// ---------------------------------------------------------------------------

typedef __attribute__((ext_vector_type(16))) __bf16 v16bf;
typedef __attribute__((ext_vector_type(8)))  __bf16 v8bf;
typedef __attribute__((ext_vector_type(8)))  float  v8f;

// Problem constants (from reference setup_inputs)
constexpr int N   = 8;
constexpr int CK  = 128;      // key channels  (= 4 WMMA k-steps of 32)
constexpr int CV  = 512;      // value channels (= 32 tiles of 16)
constexpr int Q   = 900;      // H*W queries
constexpr int QP2 = 928;      // Q padded to multiple of 32
constexpr int QT  = 57;       // q-tiles that contain at least one valid q
constexpr int QG  = QP2 / 32; // 29 q-pair groups for the output kernel
constexpr int MM  = 7200;     // T*H*W memory positions
constexpr int MT  = MM / 16;  // 450 m-tiles
constexpr int MH0 = 3584;     // m split: half 0 (112 k-steps)
constexpr int MH1 = 3616;     // m split: half 1 (113 k-steps)
constexpr float SCALE = 0.08838834764831845f; // 1/sqrt(128), folded into qk

// workspace layout (bytes)
constexpr size_t QKT_BYTES = (size_t)N * QP2 * CK * 2; //  1.90 MB  bf16 [n][q][c]
constexpr size_t MKB_BYTES = (size_t)N * CK * MM * 2;  // 14.75 MB  bf16 [n][c][m]
constexpr size_t MVB_BYTES = (size_t)N * CV * MM * 2;  // 58.98 MB  bf16 [n][c][m]
constexpr size_t CMX_BYTES = (size_t)N * MM * 4;       //  0.23 MB  f32 colmax
constexpr size_t LDS_BYTES = (size_t)MH1 * 32 * 2;     // 231424 B  P strip in LDS

__device__ inline v16bf concat8(v8bf lo, v8bf hi) {
  return __builtin_shufflevector(lo, hi, 0,1,2,3,4,5,6,7,8,9,10,11,12,13,14,15);
}

// A-fragment (16x32 bf16): lane row = M, lane-half selects K sub-ranges
// {k0..k0+7, k0+16..k0+23} (lo lanes) / {k0+8..k0+15, k0+24..k0+31} (hi lanes)
__device__ inline v16bf load_a_frag(const __bf16* rowbase, int k0, int hl) {
  v8bf lo = *(const v8bf*)(rowbase + k0 + hl * 8);
  v8bf hi = *(const v8bf*)(rowbase + k0 + 16 + hl * 8);
  return concat8(lo, hi);
}

// B-fragment (32x16 bf16): elements s = 0..15 are 16 contiguous bf16 at p
__device__ inline v16bf load_b16(const __bf16* p) {
  v8bf lo = *(const v8bf*)(p);
  v8bf hi = *(const v8bf*)(p + 8);
  return concat8(lo, hi);
}

__device__ inline v8f wmma_bf16(v16bf a, v16bf b, v8f c) {
  return __builtin_amdgcn_wmma_f32_16x16x32_bf16(
      /*neg_a=*/false, a, /*neg_b=*/false, b,
      /*c_mod=*/(short)0, c, /*reuse_a=*/false, /*reuse_b=*/false);
}

// ---------------------------------------------------------------------------
// Prep kernels: f32 -> bf16 staging (working set is L2-resident), q_val copy
// ---------------------------------------------------------------------------

__global__ void stm_prep_qkT(const float* __restrict__ qk, __bf16* __restrict__ qkT) {
  int i = blockIdx.x * 256 + threadIdx.x;            // over N*QP2*CK
  if (i >= N * QP2 * CK) return;
  int c  = i % CK;
  int t  = i / CK;
  int qp = t % QP2;
  int n  = t / QP2;
  float v = (qp < Q) ? qk[((size_t)n * CK + c) * Q + qp] * SCALE : 0.0f;
  qkT[i] = (__bf16)v;
}

__global__ void stm_cast_bf16(const float* __restrict__ src, __bf16* __restrict__ dst, int count) {
  int i = blockIdx.x * 256 + threadIdx.x;
  if (i < count) dst[i] = (__bf16)src[i];
}

__global__ void stm_copy_qval(const float* __restrict__ qv, float* __restrict__ out) {
  int i = blockIdx.x * 256 + threadIdx.x;            // over N*CV*Q
  if (i >= N * CV * Q) return;
  int q = i % Q;
  int t = i / Q;
  int c = t % CV;
  int n = t / CV;
  out[((size_t)n * 1024 + c) * Q + q] = qv[i];
}

// ---------------------------------------------------------------------------
// Kernel B: per-column (over q) online softmax stats via GEMM1 streaming.
// Block = 256 thr = 8 waves; wave owns one 16-wide m-tile, loops all q-tiles.
// ---------------------------------------------------------------------------

__global__ void stm_colstats(const __bf16* __restrict__ qkT,
                             const __bf16* __restrict__ mkB,
                             float* __restrict__ colmax,
                             float* __restrict__ colsum) {
  const int n     = blockIdx.y;
  const int lane  = threadIdx.x & 31;
  const int w     = threadIdx.x >> 5;
  const int mtile = blockIdx.x * 8 + w;
  if (mtile >= MT) return;
  const int m0 = mtile * 16;
  const int la = lane & 15;
  const int hl = lane >> 4;

  float mx = -3.0e38f, sm = 0.0f;   // stats for column m0+la, rows (q%16) in [8hl, 8hl+8)

  for (int qt = 0; qt < QT; ++qt) {
    v8f d = {0.f,0.f,0.f,0.f,0.f,0.f,0.f,0.f};
    const __bf16* arow = qkT + ((size_t)n * QP2 + qt * 16 + la) * CK;
    #pragma unroll
    for (int ks = 0; ks < 4; ++ks) {
      const int k0 = ks * 32;
      v16bf a = load_a_frag(arow, k0, hl);
      v16bf b;
      #pragma unroll
      for (int s = 0; s < 16; ++s) {
        const int kk = (s < 8) ? hl * 8 + s : 16 + hl * 8 + (s - 8);
        b[s] = mkB[((size_t)n * CK + k0 + kk) * MM + m0 + la];
      }
      d = wmma_bf16(a, b, d);
    }
    const int qbase = qt * 16 + hl * 8;
    #pragma unroll
    for (int r = 0; r < 8; ++r) {
      if (qbase + r < Q) {                       // mask padded q rows
        float v  = d[r];
        float nm = fmaxf(mx, v);
        sm = sm * __expf(mx - nm) + __expf(v - nm);
        mx = nm;
      }
    }
  }
  // combine the two lane-halves (rows 0-7 vs 8-15 of each tile)
  float mo = __shfl_xor(mx, 16);
  float so = __shfl_xor(sm, 16);
  float fm = fmaxf(mx, mo);
  float fs = sm * __expf(mx - fm) + so * __expf(mo - fm);
  if (hl == 0) {
    colmax[(size_t)n * MM + m0 + la] = fm;
    colsum[(size_t)n * MM + m0 + la] = fs;
  }
}

// ---------------------------------------------------------------------------
// Kernel C: per (n, 32-wide q group): for each m-half, recompute the P strip
// [mh x 32] into LDS (231 KB), then accumulate GEMM2 over that k-range.
// Block = 512 thr = 16 waves; wave owns 2 c-tiles x 2 q-tiles (32 acc VGPRs).
// Each mv A-fragment now feeds 2 WMMAs -> halves L2 traffic on mv.
// ---------------------------------------------------------------------------

__global__ void stm_output(const __bf16* __restrict__ qkT,
                           const __bf16* __restrict__ mkB,
                           const __bf16* __restrict__ mvB,
                           const float* __restrict__ colmax,
                           const float* __restrict__ colsum,
                           float* __restrict__ out) {
  extern __shared__ char smem_raw[];
  __bf16* P = (__bf16*)smem_raw;                 // [mh][32] row = local m, col = local q

  const int n    = blockIdx.y;
  const int q0   = blockIdx.x * 32;
  const int lane = threadIdx.x & 31;
  const int w    = threadIdx.x >> 5;             // 0..15
  const int la   = lane & 15;
  const int hl   = lane >> 4;

  v8f acc[2][2];                                 // [c-tile][q-tile]
  #pragma unroll
  for (int t = 0; t < 2; ++t)
    #pragma unroll
    for (int j = 0; j < 2; ++j)
      acc[t][j] = (v8f){0.f,0.f,0.f,0.f,0.f,0.f,0.f,0.f};

  #pragma unroll
  for (int half = 0; half < 2; ++half) {
    const int mbase  = half ? MH0 : 0;
    const int mh     = half ? MH1 : MH0;
    const int mtiles = mh / 16;

    // ---- Phase 1: P[mloc][qloc] = exp(aff - colmax[m]) / colsum[m] ----
    for (int job = w; job < mtiles * 2; job += 16) {
      const int mt  = job >> 1;                  // local m-tile
      const int j   = job & 1;                   // q-tile within the pair
      const int m0g = mbase + mt * 16;
      const __bf16* arow = qkT + ((size_t)n * QP2 + q0 + j * 16 + la) * CK;
      v8f d = {0.f,0.f,0.f,0.f,0.f,0.f,0.f,0.f};
      #pragma unroll
      for (int ks = 0; ks < 4; ++ks) {
        const int k0 = ks * 32;
        v16bf a = load_a_frag(arow, k0, hl);
        v16bf b;
        #pragma unroll
        for (int s = 0; s < 16; ++s) {
          const int kk = (s < 8) ? hl * 8 + s : 16 + hl * 8 + (s - 8);
          b[s] = mkB[((size_t)n * CK + k0 + kk) * MM + m0g + la];
        }
        d = wmma_bf16(a, b, d);
      }
      const float cm = colmax[(size_t)n * MM + m0g + la];
      const float rs = 1.0f / colsum[(size_t)n * MM + m0g + la];
      v8bf p;
      #pragma unroll
      for (int r = 0; r < 8; ++r) p[r] = (__bf16)(__expf(d[r] - cm) * rs);
      // D-frag lane holds column m = m0g+la, rows (q%16) in [8hl,8hl+8): one b128 store
      *(v8bf*)(P + ((size_t)(mt * 16 + la)) * 32 + j * 16 + hl * 8) = p;
    }
    __syncthreads();

    // ---- Phase 2: mapped[c, q] += mv[c, m] * P[m, q] over this m-half ----
    const int nks = mh / 32;
    for (int ks = 0; ks < nks; ++ks) {
      const int k0l = ks * 32;                   // local m
      const int k0g = mbase + k0l;               // global m
      // B-frags from LDS: lane = q-column la, elements = K rows m in table order
      v16bf b0, b1;
      #pragma unroll
      for (int s = 0; s < 16; ++s) {
        const int kk = (s < 8) ? hl * 8 + s : 16 + hl * 8 + (s - 8);
        b0[s] = P[((size_t)(k0l + kk)) * 32 + la];
        b1[s] = P[((size_t)(k0l + kk)) * 32 + 16 + la];
      }
      #pragma unroll
      for (int t = 0; t < 2; ++t) {
        const int c = w * 32 + t * 16 + la;
        v16bf a = load_a_frag(mvB + ((size_t)n * CV + c) * MM, k0g, hl);
        acc[t][0] = wmma_bf16(a, b0, acc[t][0]);
        acc[t][1] = wmma_bf16(a, b1, acc[t][1]);
      }
    }
    __syncthreads();                             // LDS reused by next half
  }

  // ---- store: out[n][512 + c][q], mask q >= 900 ----
  #pragma unroll
  for (int j = 0; j < 2; ++j) {
    const int qcol = q0 + j * 16 + la;
    if (qcol < Q) {
      #pragma unroll
      for (int t = 0; t < 2; ++t) {
        const int cbase = w * 32 + t * 16 + hl * 8;
        #pragma unroll
        for (int r = 0; r < 8; ++r) {
          out[((size_t)n * 1024 + 512 + cbase + r) * Q + qcol] = acc[t][j][r];
        }
      }
    }
  }
}

// ---------------------------------------------------------------------------

extern "C" void kernel_launch(void* const* d_in, const int* in_sizes, int n_in,
                              void* d_out, int out_size, void* d_ws, size_t ws_size,
                              hipStream_t stream) {
  const float* q_key = (const float*)d_in[0];
  const float* q_val = (const float*)d_in[1];
  const float* m_key = (const float*)d_in[2];
  const float* m_val = (const float*)d_in[3];
  float* out = (float*)d_out;

  char* ws = (char*)d_ws;
  __bf16* qkT    = (__bf16*)(ws);
  __bf16* mkB    = (__bf16*)(ws + QKT_BYTES);
  __bf16* mvB    = (__bf16*)(ws + QKT_BYTES + MKB_BYTES);
  float*  colmax = (float*)(ws + QKT_BYTES + MKB_BYTES + MVB_BYTES);
  float*  colsum = (float*)(ws + QKT_BYTES + MKB_BYTES + MVB_BYTES + CMX_BYTES);

  // Allow >64KB dynamic LDS for the output kernel (CDNA5 WGP has 320KB)
  (void)hipFuncSetAttribute((const void*)stm_output,
                            hipFuncAttributeMaxDynamicSharedMemorySize,
                            (int)LDS_BYTES);

  // Stage bf16 operands + copy q_val into out[:, :512, :]
  {
    int cnt = N * QP2 * CK;
    stm_prep_qkT<<<(cnt + 255) / 256, 256, 0, stream>>>(q_key, qkT);
  }
  {
    int cnt = N * CK * MM;
    stm_cast_bf16<<<(cnt + 255) / 256, 256, 0, stream>>>(m_key, mkB, cnt);
  }
  {
    int cnt = N * CV * MM;
    stm_cast_bf16<<<(cnt + 255) / 256, 256, 0, stream>>>(m_val, mvB, cnt);
  }
  {
    int cnt = N * CV * Q;
    stm_copy_qval<<<(cnt + 255) / 256, 256, 0, stream>>>(q_val, out);
  }

  // Column softmax stats (softmax is over the query axis)
  {
    dim3 grid((MT + 7) / 8, N);
    stm_colstats<<<grid, 256, 0, stream>>>(qkT, mkB, colmax, colsum);
  }

  // Recompute P into LDS + GEMM2 + store
  {
    dim3 grid(QG, N);
    stm_output<<<grid, 512, LDS_BYTES, stream>>>(qkT, mkB, mvB, colmax, colsum, out);
  }
}
